// GQAttention_41489384079983
// MI455X (gfx1250) — compile-verified, matrix operand order checked by reading, not run on hardware
//
#include <hip/hip_runtime.h>

typedef __attribute__((ext_vector_type(16))) __bf16 v16bf;
typedef __attribute__((ext_vector_type(8)))  float  v8f;

#define B_   2
#define T_   2048
#define D_   2048
#define H_   32
#define KVH_ 8
#define HD_  64
#define NREP_ (H_ / KVH_)

static __device__ __forceinline__ __bf16 f2bf(float f) { return (__bf16)f; }

// ---------------------------------------------------------------------------
// fp32 -> bf16 elementwise (vectorized x4). n must be a multiple of 4.
// ---------------------------------------------------------------------------
__global__ void cvt_bf16(const float* __restrict__ in, __bf16* __restrict__ out, int n)
{
    const int i = (blockIdx.x * blockDim.x + threadIdx.x) * 4;
    if (i >= n) return;
    const float4 v = *(const float4*)(in + i);
    out[i]     = f2bf(v.x);
    out[i + 1] = f2bf(v.y);
    out[i + 2] = f2bf(v.z);
    out[i + 3] = f2bf(v.w);
}

// ---------------------------------------------------------------------------
// fp32 [K,N] -> bf16 [N,K] tiled transpose-convert (32x32 tiles via LDS).
// Both global reads and writes fully coalesced. K,N multiples of 32.
// ---------------------------------------------------------------------------
__global__ void transpose_bf16(const float* __restrict__ in, __bf16* __restrict__ out,
                               int K, int N)
{
    __shared__ float tile[32][33];
    const int n0 = blockIdx.x * 32;
    const int k0 = blockIdx.y * 32;
    const int tx = threadIdx.x;          // 0..31
    const int ty = threadIdx.y;          // 0..7
#pragma unroll
    for (int j = 0; j < 4; ++j)
        tile[ty + 8 * j][tx] = in[(size_t)(k0 + ty + 8 * j) * N + n0 + tx];
    __syncthreads();
#pragma unroll
    for (int j = 0; j < 4; ++j)
        out[(size_t)(n0 + ty + 8 * j) * K + k0 + tx] = f2bf(tile[tx][ty + 8 * j]);
}

// ---------------------------------------------------------------------------
// Tiled NT GEMM: C[M,N] = A[M,K] @ Bt[N,K]^T, both operands bf16 K-major.
// 256 threads = 8 waves. Macro tile 64(M) x 256(N) staged in LDS (~25 KB),
// each wave register-blocks 32x64 -> 8 WMMAs per 32-wide k-step.
// OUT_MODE: 0 = f32 row-major
//           2 = bf16 head-transposed [B, NH, T, HD]   (Q/K projections)
//           3 = bf16 V-transposed    [B, NH, HD, T]   (V projection)
// ---------------------------------------------------------------------------
template <int OUT_MODE>
__global__ void gemm_nt(const __bf16* __restrict__ A,   // [M,K]
                        const __bf16* __restrict__ Bt,  // [N,K]
                        void* __restrict__ C, int M, int N, int K, int NH)
{
    __shared__ __bf16 As[64][40];    // stride 40 to spread banks
    __shared__ __bf16 Bs[256][40];

    const int tid   = threadIdx.x;
    const int lane  = tid & 31;
    const int wave  = tid >> 5;
    const int wm    = wave & 1;          // 2 waves in M -> 32-row strips
    const int wn    = wave >> 1;         // 4 waves in N -> 64-col strips
    const int m0    = blockIdx.y * 64;
    const int n0    = blockIdx.x * 256;

    const int mr    = lane & 15;
    const int kb    = (lane >> 4) * 8;
    const int rbase = (lane >> 4) * 8;

    const int lrow  = tid >> 2;          // 0..63 : cooperative-load row
    const int lc8   = (tid & 3) * 8;     // 0,8,16,24 : 8-element column chunk

    v8f acc[2][4];
#pragma unroll
    for (int mt = 0; mt < 2; ++mt)
#pragma unroll
        for (int nt = 0; nt < 4; ++nt)
#pragma unroll
            for (int i = 0; i < 8; ++i) acc[mt][nt][i] = 0.0f;

    const __bf16* aSrc = A  + (size_t)(m0 + lrow) * K + lc8;
    const __bf16* bSrc = Bt + (size_t)(n0 + lrow) * K + lc8;

    for (int k0 = 0; k0 < K; k0 += 32) {
        // ---- coalesced 16B global loads into registers ----
        const uint4 av = *(const uint4*)(aSrc + k0);
        uint4 bv[4];
#pragma unroll
        for (int j = 0; j < 4; ++j)
            bv[j] = *(const uint4*)(bSrc + (size_t)(64 * j) * K + k0);
        if (k0 + 32 < K) {
            __builtin_prefetch(aSrc + k0 + 32, 0, 1);
            __builtin_prefetch(bSrc + k0 + 32, 0, 1);
        }

        __syncthreads();   // WAR: previous iteration's fragment reads done
        *(uint4*)&As[lrow][lc8] = av;
#pragma unroll
        for (int j = 0; j < 4; ++j)
            *(uint4*)&Bs[lrow + 64 * j][lc8] = bv[j];
        __syncthreads();   // RAW: tiles visible

        // ---- build fragments (contiguous 16B LDS reads) ----
        v16bf aF[2], bF[4];
#pragma unroll
        for (int mt = 0; mt < 2; ++mt) {
            const __bf16* r = &As[wm * 32 + mt * 16 + mr][0];
#pragma unroll
            for (int i = 0; i < 8; ++i) {
                aF[mt][i]     = r[kb + i];
                aF[mt][8 + i] = r[16 + kb + i];
            }
        }
#pragma unroll
        for (int nt = 0; nt < 4; ++nt) {
            const __bf16* r = &Bs[wn * 64 + nt * 16 + mr][0];
#pragma unroll
            for (int i = 0; i < 8; ++i) {
                bF[nt][i]     = r[kb + i];
                bF[nt][8 + i] = r[16 + kb + i];
            }
        }

        // ---- 8 WMMAs per k-step ----
#pragma unroll
        for (int mt = 0; mt < 2; ++mt)
#pragma unroll
            for (int nt = 0; nt < 4; ++nt)
                acc[mt][nt] = __builtin_amdgcn_wmma_f32_16x16x32_bf16(
                    false, aF[mt], false, bF[nt], (short)0, acc[mt][nt], false, false);
    }

    // ---- epilogue ----
#pragma unroll
    for (int mt = 0; mt < 2; ++mt)
#pragma unroll
        for (int nt = 0; nt < 4; ++nt)
#pragma unroll
            for (int i = 0; i < 8; ++i) {
                const int row = m0 + wm * 32 + mt * 16 + rbase + i;
                const int col = n0 + wn * 64 + nt * 16 + mr;
                if (OUT_MODE == 0) {
                    ((float*)C)[(size_t)row * N + col] = acc[mt][nt][i];
                } else if (OUT_MODE == 2) {
                    const int bb = row / T_, t = row % T_;
                    const int h  = col / HD_, d = col % HD_;
                    ((__bf16*)C)[(((size_t)bb * NH + h) * T_ + t) * HD_ + d] =
                        f2bf(acc[mt][nt][i]);
                } else {  // V: [B, NH, HD, T]
                    const int bb = row / T_, t = row % T_;
                    const int h  = col / HD_, d = col % HD_;
                    ((__bf16*)C)[(((size_t)bb * NH + h) * HD_ + d) * T_ + t] =
                        f2bf(acc[mt][nt][i]);
                }
            }
}

// ---------------------------------------------------------------------------
// In-place RoPE on [B, NH, T, HD] bf16; one thread per (d, d+32) pair.
// `scale` folds 1/sqrt(HD) into Q.
// ---------------------------------------------------------------------------
__global__ void rope_inplace(__bf16* __restrict__ q, const float* __restrict__ cosT,
                             const float* __restrict__ sinT, int NH, float scale)
{
    const int idx   = blockIdx.x * blockDim.x + threadIdx.x;
    const int total = B_ * NH * T_ * (HD_ / 2);
    if (idx >= total) return;
    const int d    = idx & 31;
    const int rest = idx >> 5;              // (b*NH + h)*T + t
    const int t    = rest % T_;
    const size_t base = (size_t)rest * HD_;
    const float c  = cosT[t * 32 + d];
    const float s  = sinT[t * 32 + d];
    const float lo = (float)q[base + d];
    const float hi = (float)q[base + d + 32];
    q[base + d]      = f2bf((lo * c - hi * s) * scale);
    q[base + d + 32] = f2bf((hi * c + lo * s) * scale);
}

// ---------------------------------------------------------------------------
// Flash attention, causal, GQA. One wave per (b, h, 16-query tile).
// 32 keys/iter: QK^T = 4 WMMAs, online softmax in half-wave shuffles,
// P through LDS, PV = 4 WMMAs. V comes in [B,KVH,HD,T] so all operand
// fragment loads are contiguous b128.
// ---------------------------------------------------------------------------
__global__ void attn_fwd(const __bf16* __restrict__ Q,   // [B,H,T,HD]
                         const __bf16* __restrict__ Kt,  // [B,KVH,T,HD]
                         const __bf16* __restrict__ Vt,  // [B,KVH,HD,T]
                         __bf16* __restrict__ ctx)       // [B,T,H*HD]
{
    __shared__ __bf16 Plds[16][32];

    const int lane = threadIdx.x;
    const int qt   = blockIdx.x;
    const int h    = blockIdx.y;
    const int b    = blockIdx.z;
    const int kvh  = h / NREP_;
    const int q0   = qt * 16;

    const __bf16* qp = Q  + ((size_t)b * H_   + h)   * T_ * HD_;
    const __bf16* kp = Kt + ((size_t)b * KVH_ + kvh) * T_ * HD_;
    const __bf16* vp = Vt + ((size_t)b * KVH_ + kvh) * HD_ * T_;

    const int mr    = lane & 15;
    const int kb    = (lane >> 4) * 8;
    const int rbase = (lane >> 4) * 8;

    v16bf aQ[2];
#pragma unroll
    for (int kc = 0; kc < 2; ++kc) {
        const __bf16* r = qp + (size_t)(q0 + mr) * HD_ + kc * 32;
#pragma unroll
        for (int i = 0; i < 8; ++i) {
            aQ[kc][i]     = r[kb + i];
            aQ[kc][8 + i] = r[16 + kb + i];
        }
    }

    float mrow[8], lrow[8];
    v8f acc[4];
#pragma unroll
    for (int i = 0; i < 8; ++i) { mrow[i] = -1e30f; lrow[i] = 0.0f; }
#pragma unroll
    for (int nt = 0; nt < 4; ++nt)
#pragma unroll
        for (int i = 0; i < 8; ++i) acc[nt][i] = 0.0f;

    const int nchunk = (q0 + 16 + 31) / 32;
    for (int jc = 0; jc < nchunk; ++jc) {
        const int k0 = jc * 32;

        v8f s[2];
#pragma unroll
        for (int t = 0; t < 2; ++t) {
            v16bf bK[2];
            const __bf16* kr = kp + (size_t)(k0 + 16 * t + mr) * HD_;
#pragma unroll
            for (int kc = 0; kc < 2; ++kc)
#pragma unroll
                for (int i = 0; i < 8; ++i) {
                    bK[kc][i]     = kr[kc * 32 + kb + i];
                    bK[kc][8 + i] = kr[kc * 32 + 16 + kb + i];
                }
            v8f sv;
#pragma unroll
            for (int i = 0; i < 8; ++i) sv[i] = 0.0f;
            sv = __builtin_amdgcn_wmma_f32_16x16x32_bf16(false, aQ[0], false, bK[0],
                                                         (short)0, sv, false, false);
            sv = __builtin_amdgcn_wmma_f32_16x16x32_bf16(false, aQ[1], false, bK[1],
                                                         (short)0, sv, false, false);
            s[t] = sv;
        }

#pragma unroll
        for (int t = 0; t < 2; ++t) {
            const int key = k0 + 16 * t + mr;
#pragma unroll
            for (int i = 0; i < 8; ++i)
                if (key > q0 + rbase + i) s[t][i] = -1e30f;
        }

        float alpha[8];
#pragma unroll
        for (int i = 0; i < 8; ++i) {
            float rm = fmaxf(s[0][i], s[1][i]);
#pragma unroll
            for (int off = 1; off < 16; off <<= 1)
                rm = fmaxf(rm, __shfl_xor(rm, off, 32));
            const float mnew = fmaxf(mrow[i], rm);
            alpha[i] = __expf(mrow[i] - mnew);
            const float p0 = __expf(s[0][i] - mnew);
            const float p1 = __expf(s[1][i] - mnew);
            s[0][i] = p0; s[1][i] = p1;
            float r = p0 + p1;
#pragma unroll
            for (int off = 1; off < 16; off <<= 1)
                r += __shfl_xor(r, off, 32);
            lrow[i] = lrow[i] * alpha[i] + r;
            mrow[i] = mnew;
        }
#pragma unroll
        for (int nt = 0; nt < 4; ++nt)
#pragma unroll
            for (int i = 0; i < 8; ++i) acc[nt][i] *= alpha[i];

#pragma unroll
        for (int i = 0; i < 8; ++i) {
            Plds[rbase + i][mr]      = f2bf(s[0][i]);
            Plds[rbase + i][16 + mr] = f2bf(s[1][i]);
        }
        __syncthreads();
        v16bf aP;
#pragma unroll
        for (int i = 0; i < 8; ++i) {
            aP[i]     = Plds[mr][kb + i];
            aP[8 + i] = Plds[mr][16 + kb + i];
        }
        __syncthreads();

#pragma unroll
        for (int nt = 0; nt < 4; ++nt) {
            v16bf bV;
            const __bf16* vr = vp + (size_t)(nt * 16 + mr) * T_ + k0;
#pragma unroll
            for (int i = 0; i < 8; ++i) {
                bV[i]     = vr[kb + i];
                bV[8 + i] = vr[16 + kb + i];
            }
            acc[nt] = __builtin_amdgcn_wmma_f32_16x16x32_bf16(false, aP, false, bV,
                                                              (short)0, acc[nt], false, false);
        }
    }

#pragma unroll
    for (int nt = 0; nt < 4; ++nt)
#pragma unroll
        for (int i = 0; i < 8; ++i) {
            const int qrow = q0 + rbase + i;
            const int col  = h * HD_ + nt * 16 + mr;
            ctx[((size_t)b * T_ + qrow) * (H_ * HD_) + col] = f2bf(acc[nt][i] / lrow[i]);
        }
}

// ---------------------------------------------------------------------------
extern "C" void kernel_launch(void* const* d_in, const int* in_sizes, int n_in,
                              void* d_out, int out_size, void* d_ws, size_t ws_size,
                              hipStream_t stream)
{
    (void)in_sizes; (void)n_in; (void)out_size; (void)ws_size;
    const float* x    = (const float*)d_in[0];
    const float* cosT = (const float*)d_in[1];
    const float* sinT = (const float*)d_in[2];
    const float* wq   = (const float*)d_in[3];
    const float* wk   = (const float*)d_in[4];
    const float* wv   = (const float*)d_in[5];
    const float* wo   = (const float*)d_in[6];
    float* out = (float*)d_out;

    const int M = B_ * T_;                       // 4096 tokens

    // Workspace layout (bf16), ~80 MB total
    __bf16* xb   = (__bf16*)d_ws;                        // [M, D]
    __bf16* wqT  = xb   + (size_t)M * D_;                // [2048, D]
    __bf16* wkT  = wqT  + (size_t)(H_ * HD_) * D_;       // [512, D]
    __bf16* wvT  = wkT  + (size_t)(KVH_ * HD_) * D_;     // [512, D]
    __bf16* woT  = wvT  + (size_t)(KVH_ * HD_) * D_;     // [D, 2048]
    __bf16* qb   = woT  + (size_t)D_ * (H_ * HD_);       // [B,H,T,HD]
    __bf16* kbuf = qb   + (size_t)B_ * H_ * T_ * HD_;    // [B,KVH,T,HD]
    __bf16* vbuf = kbuf + (size_t)B_ * KVH_ * T_ * HD_;  // [B,KVH,HD,T]
    __bf16* ctx  = vbuf + (size_t)B_ * KVH_ * T_ * HD_;  // [B,T,H*HD]

    // ---- one-time conversions ----
    {
        const int n = M * D_;
        cvt_bf16<<<(n / 4 + 255) / 256, 256, 0, stream>>>(x, xb, n);
    }
    dim3 tb(32, 8);
    transpose_bf16<<<dim3((H_ * HD_) / 32, D_ / 32), tb, 0, stream>>>(wq, wqT, D_, H_ * HD_);
    transpose_bf16<<<dim3((KVH_ * HD_) / 32, D_ / 32), tb, 0, stream>>>(wk, wkT, D_, KVH_ * HD_);
    transpose_bf16<<<dim3((KVH_ * HD_) / 32, D_ / 32), tb, 0, stream>>>(wv, wvT, D_, KVH_ * HD_);
    transpose_bf16<<<dim3(D_ / 32, (H_ * HD_) / 32), tb, 0, stream>>>(wo, woT, H_ * HD_, D_);

    // ---- projections (tiled WMMA GEMMs) ----
    dim3 gblk(256);
    gemm_nt<2><<<dim3((H_ * HD_) / 256, M / 64), gblk, 0, stream>>>(
        xb, wqT, qb, M, H_ * HD_, D_, H_);
    gemm_nt<2><<<dim3((KVH_ * HD_) / 256, M / 64), gblk, 0, stream>>>(
        xb, wkT, kbuf, M, KVH_ * HD_, D_, KVH_);
    gemm_nt<3><<<dim3((KVH_ * HD_) / 256, M / 64), gblk, 0, stream>>>(
        xb, wvT, vbuf, M, KVH_ * HD_, D_, KVH_);

    // ---- RoPE (1/sqrt(HD) folded into Q) ----
    const int nq = B_ * H_ * T_ * (HD_ / 2);
    rope_inplace<<<(nq + 255) / 256, 256, 0, stream>>>(qb, cosT, sinT, H_, 0.125f);
    const int nk = B_ * KVH_ * T_ * (HD_ / 2);
    rope_inplace<<<(nk + 255) / 256, 256, 0, stream>>>(kbuf, cosT, sinT, KVH_, 1.0f);

    // ---- causal GQA flash attention ----
    attn_fwd<<<dim3(T_ / 16, H_, B_), dim3(32), 0, stream>>>(qb, kbuf, vbuf, ctx);

    // ---- output projection -> fp32 ----
    gemm_nt<0><<<dim3(D_ / 256, M / 64), gblk, 0, stream>>>(
        ctx, woT, out, M, D_, H_ * HD_, 0);
}
